// MultiheadAttn_10282151707382
// MI455X (gfx1250) — matrix operand order repaired and checked by reading, not compiled, not run on hardware
//
#include <hip/hip_runtime.h>
#include <hip/hip_bf16.h>

typedef __bf16 bf16_t;
typedef __attribute__((ext_vector_type(16))) __bf16 v16bf;
typedef __attribute__((ext_vector_type(8)))  __bf16 v8bf;
typedef __attribute__((ext_vector_type(4)))  __bf16 v4bf;
typedef __attribute__((ext_vector_type(8)))  float  v8f;
typedef __attribute__((ext_vector_type(4)))  unsigned int u32x4;
typedef __attribute__((ext_vector_type(8)))  int i32x8;
typedef __attribute__((ext_vector_type(4)))  int i32x4;

#define D_MODEL  1024
#define N_HEADS  16
#define HEAD_DIM 64
#define SEQ      2048
#define BATCH    2
#define M_ROWS   (BATCH*SEQ)   // 4096

#if __has_builtin(__builtin_amdgcn_tensor_load_to_lds)
#define HAVE_TDM 1
#if __has_include(<hip/amd_detail/amd_gfx1250_TDM.h>)
#define TDM_6ARG 1   // therock-10.0 headers -> 6-arg builtin
#endif
#endif

static __device__ __forceinline__ v16bf cat8(v8bf lo, v8bf hi) {
  return __builtin_shufflevector(lo, hi, 0,1,2,3,4,5,6,7,8,9,10,11,12,13,14,15);
}

static __device__ __forceinline__ v8f wmma_bf16(v16bf a, v16bf b, v8f c) {
  // D = A(16x32 bf16) * B(32x16 bf16) + C(16x16 f32)
  return __builtin_amdgcn_wmma_f32_16x16x32_bf16(false, a, false, b, (short)0, c,
                                                 false, false);
}

#if defined(HAVE_TDM)
// ---------------------------------------------------------------------------
// TDM: DMA a 32(k) x 128(rows) bf16 tile from a row-major [rows x 1024] matrix
// into LDS, padding each 64B row with 16B -> LDS row stride 80B (LDT=40 elems,
// bank-conflict free and 16B aligned).  2D tensor -> groups 2/3 zeroed.
// ---------------------------------------------------------------------------
static __device__ __forceinline__ void tdm_load_tile(unsigned lds_off,
                                                     const bf16_t* gptr,
                                                     unsigned tensor_rows) {
  unsigned long long ga = (unsigned long long)(size_t)gptr;
  u32x4 g0;
  g0.x = 1u;                                            // count=1 (valid), load
  g0.y = lds_off;                                       // LDS byte address
  g0.z = (unsigned)(ga & 0xffffffffu);                  // global_addr[31:0]
  g0.w = (unsigned)((ga >> 32) & 0x01ffffffu)           // global_addr[56:32]
       | (2u << 30);                                    // type=2 ("image")
  i32x8 g1;
  g1[0] = (1 << 16)          // data_size = 1 -> 2 bytes
        | (1 << 20)          // pad_enable
        | (3 << 22)          // pad_interval: 16 DWORDs (64B) before padding
        | (3 << 25);         // pad_amount:   4 DWORDs (16B)
  g1[1] = (int)(1024u << 16);                 // tensor_dim0 = 1024 (lo16)
  g1[2] = (int)((tensor_rows & 0xffffu) << 16);      // dim0 hi=0 | dim1 lo16
  g1[3] = (int)((tensor_rows >> 16) & 0xffffu) | (32 << 16); // dim1 hi | tile_dim0=32
  g1[4] = 128;                                 // tile_dim1 = 128, tile_dim2 = 0
  g1[5] = 1024;                                // tensor_dim0_stride lo32
  g1[6] = 0;                                   // stride hi16 | dim1_stride lo
  g1[7] = 0;
  i32x4 z4 = {0, 0, 0, 0};
#if defined(TDM_6ARG)
  i32x8 z8 = {0, 0, 0, 0, 0, 0, 0, 0};
  __builtin_amdgcn_tensor_load_to_lds(g0, g1, z4, z4, z8, 0);
#else
  __builtin_amdgcn_tensor_load_to_lds(g0, g1, z4, z4, 0);
#endif
}
#endif

// ---------------------------------------------------------------------------
// fp32 -> bf16 conversion (one-time; ~2us of HBM traffic at 23.3 TB/s)
// ---------------------------------------------------------------------------
__global__ __launch_bounds__(256)
void cvt_f32_bf16_kernel(const float* __restrict__ src, bf16_t* __restrict__ dst,
                         int n4) {
  int i = blockIdx.x * blockDim.x + threadIdx.x;
  if (i < n4) {
    float4 v = ((const float4*)src)[i];
    v4bf p = { (bf16_t)v.x, (bf16_t)v.y, (bf16_t)v.z, (bf16_t)v.w };
    ((v4bf*)dst)[i] = p;
  }
}

// ---------------------------------------------------------------------------
// Tiled bf16 WMMA GEMM with TDM double-buffered LDS staging:
//   C[M,N] = A[M,K] * W[N,K]^T + bias[N]
// Block tile 128x128, K-step 32, 8 waves (256 thr), wave tile 32x64.
// Wave 0 issues tensor_load_to_lds for tile k+1 while all waves run WMMAs on
// tile k (TENSORcnt + barrier handoff).
// MODE 0: scatter bf16 Q[B,H,S,Hd], K[B,H,S,Hd], V^T[B,H,Hd,S]  (N=3072)
// MODE 1: fp32 row-major output + bias                          (N=1024)
// ---------------------------------------------------------------------------
template<int MODE>
__global__ __launch_bounds__(256)
void gemm_wmma_kernel(const bf16_t* __restrict__ A,
                      const bf16_t* __restrict__ W,
                      const float*  __restrict__ bias,
                      float*        __restrict__ outF,
                      bf16_t*       __restrict__ qB,
                      bf16_t*       __restrict__ kB,
                      bf16_t*       __restrict__ vT,
                      unsigned                   w_rows)
{
  constexpr int K   = D_MODEL;  // 1024
  constexpr int BK  = 32;
  constexpr int NKB = K / BK;   // 32
  constexpr int LDT = 40;       // padded LDS row stride (80B)
  __shared__ bf16_t As[2][128*LDT];
  __shared__ bf16_t Bs[2][128*LDT];

  const int tid  = threadIdx.x;
  const int lane = tid & 31;
  const int wv   = tid >> 5;
  const int wm   = wv >> 1;     // 0..3 (M)
  const int wn   = wv & 1;      // 0..1 (N)
  const int l15  = lane & 15;
  const int half = lane >> 4;
  const int m0   = blockIdx.y * 128;
  const int n0   = blockIdx.x * 128;

  const bf16_t* Abase = A + (size_t)m0 * K;
  const bf16_t* Wbase = W + (size_t)n0 * K;

  v8f cc[2][4];
  #pragma unroll
  for (int i = 0; i < 2; ++i)
    #pragma unroll
    for (int j = 0; j < 4; ++j)
      cc[i][j] = 0.0f;

#if defined(HAVE_TDM)
  if (wv == 0) {
    tdm_load_tile((unsigned)(size_t)&As[0][0], Abase, (unsigned)M_ROWS);
    tdm_load_tile((unsigned)(size_t)&Bs[0][0], Wbase, w_rows);
  }
#endif

  for (int kbi = 0; kbi < NKB; ++kbi) {
    const int p = kbi & 1;
#if defined(HAVE_TDM)
    if (wv == 0) __builtin_amdgcn_s_wait_tensorcnt(0);   // tile kbi landed
    __syncthreads();                                     // visible to all waves
    if (wv == 0 && (kbi + 1) < NKB) {                    // prefetch tile kbi+1
      tdm_load_tile((unsigned)(size_t)&As[p ^ 1][0],
                    Abase + (kbi + 1) * BK, (unsigned)M_ROWS);
      tdm_load_tile((unsigned)(size_t)&Bs[p ^ 1][0],
                    Wbase + (kbi + 1) * BK, w_rows);
    }
#else
    // fallback staging: plain loads (both operands already bf16)
    {
      const int kb  = kbi * BK;
      const int row = tid >> 1;
      const int c   = (tid & 1) * 16;
      const float4* sa = (const float4*)(&A[(size_t)(m0+row)*K + kb + c]);
      float4 a0 = sa[0], a1 = sa[1];
      float4* da = (float4*)(&As[p][row*LDT + c]);
      da[0] = a0; da[1] = a1;
      const float4* sb = (const float4*)(&W[(size_t)(n0+row)*K + kb + c]);
      float4 b0 = sb[0], b1 = sb[1];
      float4* db = (float4*)(&Bs[p][row*LDT + c]);
      db[0] = b0; db[1] = b1;
    }
    __syncthreads();
#endif

    // A fragments: lane m=l15; half0 k=0..15, half1 k=16..31 (contiguous)
    v16bf aF[2];
    #pragma unroll
    for (int mi = 0; mi < 2; ++mi) {
      const bf16_t* base = &As[p][(wm*32 + mi*16 + l15)*LDT];
      v8bf lo = *(const v8bf*)(base + half*16);
      v8bf hi = *(const v8bf*)(base + half*16 + 8);
      aF[mi] = cat8(lo, hi);
    }
    // B fragments: lane n=l15; half0 k=0..15, half1 k=16..31 (contiguous)
    v16bf bF[4];
    #pragma unroll
    for (int ni = 0; ni < 4; ++ni)
      bF[ni] = *(const v16bf*)(&Bs[p][(wn*64 + ni*16 + l15)*LDT + half*16]);

    #pragma unroll
    for (int mi = 0; mi < 2; ++mi)
      #pragma unroll
      for (int ni = 0; ni < 4; ++ni)
        cc[mi][ni] = wmma_bf16(aF[mi], bF[ni], cc[mi][ni]);

    __syncthreads();   // all reads of buffer p done before TDM overwrites it
  }

  // ---- epilogue: C layout row = r + 8*half, col = l15 within each 16x16 tile
  #pragma unroll
  for (int mi = 0; mi < 2; ++mi) {
    #pragma unroll
    for (int ni = 0; ni < 4; ++ni) {
      const int   gn = n0 + wn*64 + ni*16 + l15;
      const float bv = bias[gn];
      #pragma unroll
      for (int r = 0; r < 8; ++r) {
        const int gm = m0 + wm*32 + mi*16 + r + 8*half;
        float v = cc[mi][ni][r] + bv;
        if constexpr (MODE == 0) {
          int b   = gm >> 11;
          int s2  = gm & (SEQ-1);
          int t   = gn >> 10;          // 0=Q 1=K 2=V
          int rem = gn & (D_MODEL-1);
          int h   = rem >> 6;
          int hd  = rem & (HEAD_DIM-1);
          size_t bh = (size_t)(b*N_HEADS + h);
          bf16_t bb = (bf16_t)v;
          if (t == 0)       qB[(bh*SEQ + s2)*HEAD_DIM + hd] = bb;
          else if (t == 1)  kB[(bh*SEQ + s2)*HEAD_DIM + hd] = bb;
          else              vT[(bh*HEAD_DIM + hd)*SEQ + s2] = bb;  // V transposed
        } else {
          outF[(size_t)gm*D_MODEL + gn] = v;
        }
      }
    }
  }
}

// ---------------------------------------------------------------------------
// Flash attention tile body.  MASK=false for tiles fully below the diagonal
// (no compares / cndmask at all), MASK=true for the single diagonal tile.
// Scale 1/sqrt(64)=0.125 is pre-folded into the Q fragment (exact in bf16).
// ---------------------------------------------------------------------------
template<bool MASK>
static __device__ __forceinline__ void attn_tile(
    int kb, int q0, int l15, int half,
    const bf16_t* __restrict__ Kp, const bf16_t* __restrict__ Vp,
    const v16bf* qF, bf16_t* __restrict__ Pw,
    v8f* acc, float* mrow, float* lrow)
{
  constexpr int LDP = 40;

  v8f sc[2];
  #pragma unroll
  for (int n0 = 0; n0 < 2; ++n0) {
    v8f c = 0.0f;
    const int key = kb + n0*16 + l15;
    #pragma unroll
    for (int s = 0; s < 2; ++s) {
      v16bf kf = *(const v16bf*)(Kp + (size_t)key*HEAD_DIM + s*32 + half*16);
      c = wmma_bf16(qF[s], kf, c);
    }
    sc[n0] = c;
  }

  if constexpr (MASK) {
    #pragma unroll
    for (int n0 = 0; n0 < 2; ++n0) {
      const int key = kb + n0*16 + l15;
      #pragma unroll
      for (int r = 0; r < 8; ++r) {
        const int q = q0 + r + 8*half;
        sc[n0][r] = (key <= q) ? sc[n0][r] : -1e30f;
      }
    }
  }

  #pragma unroll
  for (int r = 0; r < 8; ++r) {
    float t = fmaxf(sc[0][r], sc[1][r]);
    t = fmaxf(t, __shfl_xor(t, 1));
    t = fmaxf(t, __shfl_xor(t, 2));
    t = fmaxf(t, __shfl_xor(t, 4));
    t = fmaxf(t, __shfl_xor(t, 8));
    const float mn = fmaxf(mrow[r], t);
    const float rs = __expf(mrow[r] - mn);
    const float p0 = __expf(sc[0][r] - mn);
    const float p1 = __expf(sc[1][r] - mn);
    float ps = p0 + p1;
    ps += __shfl_xor(ps, 1);
    ps += __shfl_xor(ps, 2);
    ps += __shfl_xor(ps, 4);
    ps += __shfl_xor(ps, 8);
    lrow[r] = lrow[r]*rs + ps;
    mrow[r] = mn;
    #pragma unroll
    for (int n = 0; n < 4; ++n) acc[n][r] *= rs;
    const int ql = r + 8*half;
    Pw[ql*LDP + l15]      = (bf16_t)p0;
    Pw[ql*LDP + 16 + l15] = (bf16_t)p1;
  }
  asm volatile("s_wait_dscnt 0" ::: "memory");   // LDS RAW within wave

  const bf16_t* pr = &Pw[l15*LDP];
  v8bf lo = *(const v8bf*)(pr + half*16);
  v8bf hi = *(const v8bf*)(pr + half*16 + 8);
  v16bf pF = cat8(lo, hi);
  #pragma unroll
  for (int n = 0; n < 4; ++n) {
    v16bf vf = *(const v16bf*)(Vp + (size_t)(n*16 + l15)*SEQ + kb + half*16);
    acc[n] = wmma_bf16(pF, vf, acc[n]);
  }
}

// ---------------------------------------------------------------------------
// Flash attention: one wave = 16 Q rows, k-tiles of 32 keys, causal.
// (q0+1)>>5 unmasked tiles + exactly one masked diagonal tile.
// ---------------------------------------------------------------------------
__global__ __launch_bounds__(128)
void flash_attn_kernel(const bf16_t* __restrict__ qB,
                       const bf16_t* __restrict__ kB,
                       const bf16_t* __restrict__ vT,
                       bf16_t*       __restrict__ aB)
{
  constexpr int LDP = 40;
  __shared__ bf16_t P[4][16*LDP];   // per-wave P staging (C-layout -> A-layout)

  const int tid  = threadIdx.x;
  const int lane = tid & 31;
  const int wv   = tid >> 5;
  const int l15  = lane & 15;
  const int half = lane >> 4;
  const int qblk = blockIdx.x & 31;     // S/64 q-blocks
  const int bh   = blockIdx.x >> 5;     // b*H + h
  const int q0   = qblk*64 + wv*16;

  const bf16_t* Qp = qB + (size_t)bh * SEQ * HEAD_DIM;
  const bf16_t* Kp = kB + (size_t)bh * SEQ * HEAD_DIM;
  const bf16_t* Vp = vT + (size_t)bh * HEAD_DIM * SEQ;

  // Q fragments, pre-scaled by 1/sqrt(Hd) = 0.125 (exact power of two)
  v16bf qF[2];
  #pragma unroll
  for (int s = 0; s < 2; ++s) {
    const bf16_t* r = Qp + (size_t)(q0 + l15)*HEAD_DIM + s*32;
    v8bf lo = *(const v8bf*)(r + half*16);
    v8bf hi = *(const v8bf*)(r + half*16 + 8);
    v16bf q = cat8(lo, hi);
    #pragma unroll
    for (int i = 0; i < 16; ++i)
      q[i] = (bf16_t)((float)q[i] * 0.125f);
    qF[s] = q;
  }

  v8f acc[4];
  #pragma unroll
  for (int n = 0; n < 4; ++n) acc[n] = 0.0f;
  float mrow[8], lrow[8];
  #pragma unroll
  for (int r = 0; r < 8; ++r) { mrow[r] = -1e30f; lrow[r] = 0.0f; }

  const int nfull = (q0 + 1) >> 5;        // tiles fully below the diagonal
  for (int kt = 0; kt < nfull; ++kt)
    attn_tile<false>(kt*32, q0, l15, half, Kp, Vp, qF, &P[wv][0],
                     acc, mrow, lrow);
  attn_tile<true>(nfull*32, q0, l15, half, Kp, Vp, qF, &P[wv][0],
                  acc, mrow, lrow);      // the single diagonal tile

  const int b = bh >> 4;
  const int h = bh & 15;
  #pragma unroll
  for (int r = 0; r < 8; ++r) {
    const float inv = 1.0f / lrow[r];
    const int q = q0 + r + 8*half;
    bf16_t* dst = aB + ((size_t)(b*SEQ + q))*D_MODEL + h*HEAD_DIM;
    #pragma unroll
    for (int n = 0; n < 4; ++n)
      dst[n*16 + l15] = (bf16_t)(acc[n][r] * inv);
  }
}

extern "C" void kernel_launch(void* const* d_in, const int* in_sizes, int n_in,
                              void* d_out, int out_size, void* d_ws, size_t ws_size,
                              hipStream_t stream) {
  (void)in_sizes; (void)n_in; (void)out_size; (void)ws_size;
  const float* x     = (const float*)d_in[0];
  const float* qkv_w = (const float*)d_in[1];
  const float* qkv_b = (const float*)d_in[2];
  const float* out_w = (const float*)d_in[3];
  const float* out_b = (const float*)d_in[4];
  float* out = (float*)d_out;

  const size_t per = (size_t)BATCH * N_HEADS * SEQ * HEAD_DIM;  // 4M elems
  bf16_t* qB = (bf16_t*)d_ws;          // [B,H,S,Hd]
  bf16_t* kB = qB + per;               // [B,H,S,Hd]
  bf16_t* vT = kB + per;               // [B,H,Hd,S]
  bf16_t* aB = vT + per;               // [B,S,D]
  bf16_t* xb = aB + per;               // [B*S, D] bf16 copy of x
  bf16_t* wq = xb + per;               // [3D, D]  bf16 qkv_w
  bf16_t* wo = wq + (size_t)3*D_MODEL*D_MODEL;  // [D, D] bf16 out_w

  // 0) one-time fp32 -> bf16 conversions (cheap: ~50MB total)
  {
    int n4x = (M_ROWS*D_MODEL)/4;
    cvt_f32_bf16_kernel<<<dim3((n4x+255)/256), dim3(256), 0, stream>>>(x, xb, n4x);
    int n4q = (3*D_MODEL*D_MODEL)/4;
    cvt_f32_bf16_kernel<<<dim3((n4q+255)/256), dim3(256), 0, stream>>>(qkv_w, wq, n4q);
    int n4o = (D_MODEL*D_MODEL)/4;
    cvt_f32_bf16_kernel<<<dim3((n4o+255)/256), dim3(256), 0, stream>>>(out_w, wo, n4o);
  }

  // 1) QKV projection (TDM-staged bf16 WMMA GEMM, bias, scatter Q/K/V^T)
  gemm_wmma_kernel<0><<<dim3(3*D_MODEL/128, M_ROWS/128), dim3(256), 0, stream>>>(
      xb, wq, qkv_b, nullptr, qB, kB, vT, 3*D_MODEL);

  // 2) causal flash attention
  flash_attn_kernel<<<dim3(BATCH*N_HEADS*(SEQ/64)), dim3(128), 0, stream>>>(
      qB, kB, vT, aB);

  // 3) output projection (TDM-staged bf16 WMMA GEMM, bias, fp32 out)
  gemm_wmma_kernel<1><<<dim3(D_MODEL/128, M_ROWS/128), dim3(256), 0, stream>>>(
      aB, wo, out_b, out, nullptr, nullptr, nullptr, D_MODEL);
}
